// ClipLossAcc_54692113547585
// MI455X (gfx1250) — matrix operand order repaired and checked by reading, not compiled, not run on hardware
//
#include <hip/hip_runtime.h>

// ---------------------------------------------------------------------------
// CLIP contrastive loss, N=16384, D=512, fp32 in -> scalar fp32 loss.
//   Pass 0: pack IMG/TXT fp32 -> f16 in WMMA-fragment-native tiled layout (ws)
//   Pass 1: WMMA GEMM, row/col max + diagonal
//   Pass 2: WMMA GEMM, row/col sum of exp(v - max)
//   Pass 3: scalar loss reduction
// Logits (1 GB) never materialized; packed operands (32 MB) are L2-resident.
// ---------------------------------------------------------------------------

typedef __attribute__((ext_vector_type(16))) _Float16 v16h;
typedef __attribute__((ext_vector_type(8)))  float    v8f;

constexpr int D  = 512;        // feature dim (K)
constexpr int KT = D / 32;     // K-tiles per row block (16)
constexpr int WTM = 32;        // wave tile rows (2 A-tiles)
constexpr int WTN = 64;        // wave tile cols (4 B-tiles)
constexpr int TM = 128;        // WG tile rows (4 wave-rows)
constexpr int TN = 128;        // WG tile cols (2 wave-cols)

// Monotone order-preserving map float -> uint32 so integer atomicMax works.
__device__ __forceinline__ unsigned fmap(float f) {
  unsigned u = __float_as_uint(f);
  return (u & 0x80000000u) ? ~u : (u | 0x80000000u);
}
__device__ __forceinline__ float funmap(unsigned x) {
  return (x & 0x80000000u) ? __uint_as_float(x & 0x7FFFFFFFu)
                           : __uint_as_float(~x);
}

// ---------------------------------------------------------------------------
// init workspace stats
// ---------------------------------------------------------------------------
__global__ void clip_init_ws(unsigned* __restrict__ rowMaxU,
                             unsigned* __restrict__ colMaxU,
                             float* __restrict__ rowSum,
                             float* __restrict__ colSum,
                             float* __restrict__ diag, int n) {
  int i = blockIdx.x * blockDim.x + threadIdx.x;
  if (i < n) {
    rowMaxU[i] = 0u;   // fmap(x) > 0 for every real x, so 0 acts as -inf
    colMaxU[i] = 0u;
    rowSum[i]  = 0.0f;
    colSum[i]  = 0.0f;
    diag[i]    = 0.0f;
  }
}

// ---------------------------------------------------------------------------
// Pack kernels: fp32 [n,512] row-major -> f16 tiles of 16 rows x 32 K (1 KB),
// tile t = rowTile*KT + kTile, interior = lane-major WMMA fragment order.
// One thread produces one lane-chunk (16 halves = 32 B contiguous store).
// ---------------------------------------------------------------------------
__global__ __launch_bounds__(256)
void clip_pack_a(const float* __restrict__ src, _Float16* __restrict__ dst,
                 int n) {
  int t    = blockIdx.x * 256 + threadIdx.x;
  int lane = t & 31;
  int tile = t >> 5;                  // rowTile*KT + kTile
  int kt   = tile & (KT - 1);
  int rt   = tile >> 4;               // KT == 16
  if (rt >= n / 16) return;
  int lm = lane & 15, hi = lane >> 4;
  // A-matrix 16x32 layout: half 0 -> K {0..15}, half 1 -> K {16..31} (contig)
  const float* s = src + (long)(rt * 16 + lm) * D + kt * 32 + hi * 16;
  v16h v;
#pragma unroll
  for (int i = 0; i < 8; ++i) v[i]     = (_Float16)s[i];
#pragma unroll
  for (int i = 0; i < 8; ++i) v[8 + i] = (_Float16)s[8 + i];
  *(v16h*)(dst + (long)tile * 512 + lane * 16) = v;
}

__global__ __launch_bounds__(256)
void clip_pack_b(const float* __restrict__ src, _Float16* __restrict__ dst,
                 int n) {
  int t    = blockIdx.x * 256 + threadIdx.x;
  int lane = t & 31;
  int tile = t >> 5;
  int kt   = tile & (KT - 1);
  int rt   = tile >> 4;
  if (rt >= n / 16) return;
  int lm = lane & 15, hi = lane >> 4;
  // B-matrix 32x16 layout: half 0 -> K {0..15}, half 1 -> K {16..31} (contig)
  const float* s = src + (long)(rt * 16 + lm) * D + kt * 32 + hi * 16;
  v16h v;
#pragma unroll
  for (int i = 0; i < 16; ++i) v[i] = (_Float16)s[i];
  *(v16h*)(dst + (long)tile * 512 + lane * 16) = v;
}

// ---------------------------------------------------------------------------
// GEMM pass over packed f16 tiles. Wave computes 32x64; WG = 128x128.
// MAXPASS=true : row/col max + diagonal.  MAXPASS=false : sum of exp(v-max).
// ---------------------------------------------------------------------------
template <bool MAXPASS>
__global__ __launch_bounds__(256)
void clip_gemm_pass(const _Float16* __restrict__ PA,
                    const _Float16* __restrict__ PB,
                    unsigned* __restrict__ rowMaxU,
                    unsigned* __restrict__ colMaxU,
                    float* __restrict__ rowSum,
                    float* __restrict__ colSum,
                    float* __restrict__ diag, int n) {
  const int lane = threadIdx.x & 31;
  const int wave = threadIdx.x >> 5;
  const int wr   = wave & 3;    // wave row 0..3
  const int wc   = wave >> 2;   // wave col 0..1
  const int lm   = lane & 15;
  const int hi   = lane >> 4;

  const int m0 = blockIdx.y * TM + wr * WTM;   // 32-row strip base
  const int n0 = blockIdx.x * TN + wc * WTN;   // 64-col strip base
  const int mt = m0 >> 4;                      // A row-tile base (2 tiles)
  const int nt = n0 >> 4;                      // B row-tile base (4 tiles)

  // Per-fragment base pointers: each wave reads one contiguous 1 KB tile;
  // this lane's 32-byte slice:
  const _Float16* Ap0 = PA + ((long)(mt + 0) * KT) * 512 + lane * 16;
  const _Float16* Ap1 = PA + ((long)(mt + 1) * KT) * 512 + lane * 16;
  const _Float16* Bp0 = PB + ((long)(nt + 0) * KT) * 512 + lane * 16;
  const _Float16* Bp1 = PB + ((long)(nt + 1) * KT) * 512 + lane * 16;
  const _Float16* Bp2 = PB + ((long)(nt + 2) * KT) * 512 + lane * 16;
  const _Float16* Bp3 = PB + ((long)(nt + 3) * KT) * 512 + lane * 16;

  v8f acc[2][4] = {};

#pragma unroll 2
  for (int kt = 0; kt < KT; ++kt) {
    const long o = (long)kt * 512;
    // prefetch two K-tiles ahead (global_prefetch_b8)
    __builtin_prefetch(Ap0 + o + 1024, 0, 3);
    __builtin_prefetch(Bp0 + o + 1024, 0, 3);

    v16h a0 = *(const v16h*)(Ap0 + o);
    v16h a1 = *(const v16h*)(Ap1 + o);
    v16h b0 = *(const v16h*)(Bp0 + o);
    v16h b1 = *(const v16h*)(Bp1 + o);
    v16h b2 = *(const v16h*)(Bp2 + o);
    v16h b3 = *(const v16h*)(Bp3 + o);

    acc[0][0] = __builtin_amdgcn_wmma_f32_16x16x32_f16(false, a0, false, b0,
                                                       (short)0, acc[0][0], false, false);
    acc[0][1] = __builtin_amdgcn_wmma_f32_16x16x32_f16(false, a0, false, b1,
                                                       (short)0, acc[0][1], false, false);
    acc[0][2] = __builtin_amdgcn_wmma_f32_16x16x32_f16(false, a0, false, b2,
                                                       (short)0, acc[0][2], false, false);
    acc[0][3] = __builtin_amdgcn_wmma_f32_16x16x32_f16(false, a0, false, b3,
                                                       (short)0, acc[0][3], false, false);
    acc[1][0] = __builtin_amdgcn_wmma_f32_16x16x32_f16(false, a1, false, b0,
                                                       (short)0, acc[1][0], false, false);
    acc[1][1] = __builtin_amdgcn_wmma_f32_16x16x32_f16(false, a1, false, b1,
                                                       (short)0, acc[1][1], false, false);
    acc[1][2] = __builtin_amdgcn_wmma_f32_16x16x32_f16(false, a1, false, b2,
                                                       (short)0, acc[1][2], false, false);
    acc[1][3] = __builtin_amdgcn_wmma_f32_16x16x32_f16(false, a1, false, b3,
                                                       (short)0, acc[1][3], false, false);
  }

  // acc[r][a] element e (this lane) = logits[m0 + r*16 + hi*8 + e][n0 + a*16 + lm]

  if (MAXPASS) {
    // per-row max over this wave's 64 columns
#pragma unroll
    for (int r = 0; r < 2; ++r)
#pragma unroll
      for (int e = 0; e < 8; ++e) {
        float v = fmaxf(fmaxf(acc[r][0][e], acc[r][1][e]),
                        fmaxf(acc[r][2][e], acc[r][3][e]));
#pragma unroll
        for (int mask = 1; mask < 16; mask <<= 1)
          v = fmaxf(v, __shfl_xor(v, mask, 32));
        if (lm == 0) atomicMax(&rowMaxU[m0 + r * 16 + hi * 8 + e], fmap(v));
      }
    // per-col max over this wave's 32 rows
#pragma unroll
    for (int a = 0; a < 4; ++a) {
      float v = acc[0][a][0];
#pragma unroll
      for (int r = 0; r < 2; ++r)
#pragma unroll
        for (int e = 0; e < 8; ++e) v = fmaxf(v, acc[r][a][e]);
      v = fmaxf(v, __shfl_xor(v, 16, 32));
      if (hi == 0) atomicMax(&colMaxU[n0 + a * 16 + lm], fmap(v));
    }
    // diagonal capture (only tiles straddling the diagonal)
    if (m0 < n0 + WTN && n0 < m0 + WTM) {
#pragma unroll
      for (int r = 0; r < 2; ++r)
#pragma unroll
        for (int a = 0; a < 4; ++a)
#pragma unroll
          for (int e = 0; e < 8; ++e) {
            int gm = m0 + r * 16 + hi * 8 + e;
            int gn = n0 + a * 16 + lm;
            if (gm == gn) diag[gm] = acc[r][a][e];
          }
    }
  } else {
    float rM[2][8];
#pragma unroll
    for (int r = 0; r < 2; ++r)
#pragma unroll
      for (int e = 0; e < 8; ++e)
        rM[r][e] = funmap(rowMaxU[m0 + r * 16 + hi * 8 + e]);
    float cM[4];
#pragma unroll
    for (int a = 0; a < 4; ++a) cM[a] = funmap(colMaxU[n0 + a * 16 + lm]);

    float rs[2][8] = {};
    float cs[4]    = {};
#pragma unroll
    for (int r = 0; r < 2; ++r)
#pragma unroll
      for (int a = 0; a < 4; ++a)
#pragma unroll
        for (int e = 0; e < 8; ++e) {
          float v = acc[r][a][e];
          rs[r][e] += __expf(v - rM[r][e]);
          cs[a]    += __expf(v - cM[a]);
        }
#pragma unroll
    for (int r = 0; r < 2; ++r)
#pragma unroll
      for (int e = 0; e < 8; ++e) {
        float v = rs[r][e];
#pragma unroll
        for (int mask = 1; mask < 16; mask <<= 1) v += __shfl_xor(v, mask, 32);
        if (lm == 0) atomicAdd(&rowSum[m0 + r * 16 + hi * 8 + e], v);
      }
#pragma unroll
    for (int a = 0; a < 4; ++a) {
      float v = cs[a] + __shfl_xor(cs[a], 16, 32);
      if (hi == 0) atomicAdd(&colSum[n0 + a * 16 + lm], v);
    }
  }
}

// ---------------------------------------------------------------------------
// loss = mean_i [ 0.5*(log(rowSum_i)+rowMax_i + log(colSum_i)+colMax_i) - diag_i ]
// ---------------------------------------------------------------------------
__global__ __launch_bounds__(256)
void clip_finalize(const unsigned* __restrict__ rowMaxU,
                   const unsigned* __restrict__ colMaxU,
                   const float* __restrict__ rowSum,
                   const float* __restrict__ colSum,
                   const float* __restrict__ diag,
                   float* __restrict__ out, int n) {
  __shared__ float smem[256];
  float accv = 0.0f;
  for (int i = threadIdx.x; i < n; i += 256) {
    float lr = __logf(rowSum[i]) + funmap(rowMaxU[i]);
    float lc = __logf(colSum[i]) + funmap(colMaxU[i]);
    accv += 0.5f * (lr + lc) - diag[i];
  }
  smem[threadIdx.x] = accv;
  __syncthreads();
  for (int o = 128; o > 0; o >>= 1) {
    if ((int)threadIdx.x < o) smem[threadIdx.x] += smem[threadIdx.x + o];
    __syncthreads();
  }
  if (threadIdx.x == 0) out[0] = smem[0] / (float)n;
}

// ---------------------------------------------------------------------------
extern "C" void kernel_launch(void* const* d_in, const int* in_sizes, int n_in,
                              void* d_out, int out_size, void* d_ws, size_t ws_size,
                              hipStream_t stream) {
  const float* IMG = (const float*)d_in[0];
  const float* TXT = (const float*)d_in[1];
  float* out = (float*)d_out;

  const int n = in_sizes[0] / D;  // 16384

  // workspace: [stats: 5*n*4 = 320 KB][packA: n*D f16 = 16 MB][packB: 16 MB]
  unsigned* rowMaxU = (unsigned*)d_ws;
  unsigned* colMaxU = rowMaxU + n;
  float* rowSum = (float*)(colMaxU + n);
  float* colSum = rowSum + n;
  float* diag   = colSum + n;
  _Float16* PA  = (_Float16*)((char*)d_ws + (size_t)5 * n * 4);
  _Float16* PB  = PA + (size_t)n * D;

  clip_init_ws<<<(n + 255) / 256, 256, 0, stream>>>(rowMaxU, colMaxU, rowSum,
                                                    colSum, diag, n);

  const int packThreads = (n / 16) * KT * 32;  // one thread per 16-half chunk
  clip_pack_a<<<packThreads / 256, 256, 0, stream>>>(IMG, PA, n);
  clip_pack_b<<<packThreads / 256, 256, 0, stream>>>(TXT, PB, n);

  dim3 grid(n / TN, n / TM);  // 128 x 128 workgroups
  clip_gemm_pass<true><<<grid, 256, 0, stream>>>(PA, PB, rowMaxU, colMaxU,
                                                 rowSum, colSum, diag, n);
  clip_gemm_pass<false><<<grid, 256, 0, stream>>>(PA, PB, rowMaxU, colMaxU,
                                                  rowSum, colSum, diag, n);

  clip_finalize<<<1, 256, 0, stream>>>(rowMaxU, colMaxU, rowSum, colSum, diag,
                                       out, n);
}
